// Sliding_Full_connection_Network_38903813767382
// MI455X (gfx1250) — matrix operand order repaired and checked by reading, not compiled, not run on hardware
//
#include <hip/hip_runtime.h>
#include <math.h>

typedef __attribute__((ext_vector_type(16))) _Float16 v16h;
typedef __attribute__((ext_vector_type(8)))  _Float16 v8h;
typedef __attribute__((ext_vector_type(8)))  float    v8f;

// ---------------------------------------------------------------------------
// Fold BN (gamma,beta,mean,var) and optional conv bias into per-channel
// scale/bias:  y = conv*scale + bias
// ---------------------------------------------------------------------------
__global__ void fold_bn_kernel(const float* __restrict__ g, const float* __restrict__ b,
                               const float* __restrict__ m, const float* __restrict__ v,
                               const float* __restrict__ convb,
                               float* __restrict__ scale, float* __restrict__ bias, int C)
{
  int c = blockIdx.x * blockDim.x + threadIdx.x;
  if (c < C) {
    float s = g[c] * rsqrtf(v[c] + 1e-5f);
    float bb = b[c] - m[c] * s;
    if (convb) bb += convb[c] * s;
    scale[c] = s;
    bias[c]  = bb;
  }
}

// ---------------------------------------------------------------------------
// Repack OIHW fp32 weights -> [Cout][KH][KW][Cin] fp16 (K contiguous, matching
// the NHWC im2col patch order).
// ---------------------------------------------------------------------------
__global__ void repack_w_kernel(const float* __restrict__ w, _Float16* __restrict__ o,
                                int Cout, int Cin, int KH, int KW)
{
  long idx = (long)blockIdx.x * blockDim.x + threadIdx.x;
  long total = (long)Cout * Cin * KH * KW;
  if (idx >= total) return;
  int cin = (int)(idx % Cin); long t = idx / Cin;
  int kw  = (int)(t % KW);    t /= KW;
  int kh  = (int)(t % KH);
  int co  = (int)(t / KH);
  o[idx] = (_Float16)w[(((long)co * Cin + cin) * KH + kh) * KW + kw];
}

// ---------------------------------------------------------------------------
// Stem: 7x7 s2 p3 conv (Cin=3) + BN + ReLU.  NCHW fp32 in -> NHWC fp16 out.
// ---------------------------------------------------------------------------
__global__ void stem_kernel(const float* __restrict__ x, const float* __restrict__ w,
                            const float* __restrict__ scale, const float* __restrict__ bias,
                            _Float16* __restrict__ out)
{
  long idx = (long)blockIdx.x * blockDim.x + threadIdx.x;
  const long total = (long)8 * 112 * 112 * 64;
  if (idx >= total) return;
  int c  = (int)(idx & 63); long t = idx >> 6;
  int wo = (int)(t % 112);  t /= 112;
  int ho = (int)(t % 112);
  int n  = (int)(t / 112);
  float acc = 0.f;
  for (int ci = 0; ci < 3; ++ci)
    for (int kh = 0; kh < 7; ++kh) {
      int hi = ho * 2 - 3 + kh;
      if (hi < 0 || hi >= 224) continue;
      for (int kw = 0; kw < 7; ++kw) {
        int wi = wo * 2 - 3 + kw;
        if (wi < 0 || wi >= 224) continue;
        acc += x[(((long)n * 3 + ci) * 224 + hi) * 224 + wi] *
               w[(((long)c * 3 + ci) * 7 + kh) * 7 + kw];
      }
    }
  float vv = acc * scale[c] + bias[c];
  out[idx] = (_Float16)(vv > 0.f ? vv : 0.f);
}

// ---------------------------------------------------------------------------
// Maxpool 3x3 s2 p1, NHWC fp16: [8,112,112,64] -> [8,56,56,64]
// ---------------------------------------------------------------------------
__global__ void maxpool_kernel(const _Float16* __restrict__ in, _Float16* __restrict__ out)
{
  long idx = (long)blockIdx.x * blockDim.x + threadIdx.x;
  const long total = (long)8 * 56 * 56 * 64;
  if (idx >= total) return;
  int c  = (int)(idx & 63); long t = idx >> 6;
  int wo = (int)(t % 56);   t /= 56;
  int ho = (int)(t % 56);
  int n  = (int)(t / 56);
  float best = -3.4e38f;
  for (int kh = 0; kh < 3; ++kh) {
    int hi = ho * 2 - 1 + kh;
    if (hi < 0 || hi >= 112) continue;
    for (int kw = 0; kw < 3; ++kw) {
      int wi = wo * 2 - 1 + kw;
      if (wi < 0 || wi >= 112) continue;
      float v = (float)in[(((long)n * 112 + hi) * 112 + wi) * 64 + c];
      best = v > best ? v : best;
    }
  }
  out[idx] = (_Float16)best;
}

// ---------------------------------------------------------------------------
// Elementwise add (stage-level shortcut), fp16
// ---------------------------------------------------------------------------
__global__ void add_kernel(const _Float16* __restrict__ a, const _Float16* __restrict__ b,
                           _Float16* __restrict__ o, long n)
{
  long i = (long)blockIdx.x * blockDim.x + threadIdx.x;
  if (i < n) o[i] = (_Float16)((float)a[i] + (float)b[i]);
}

// ---------------------------------------------------------------------------
// Global average pool: [8,7,7,2048] fp16 -> [8,2048] fp32
// ---------------------------------------------------------------------------
__global__ void avgpool_kernel(const _Float16* __restrict__ in, float* __restrict__ out)
{
  int idx = blockIdx.x * blockDim.x + threadIdx.x;
  if (idx >= 8 * 2048) return;
  int c = idx & 2047;
  int n = idx >> 11;
  float s = 0.f;
  for (int h = 0; h < 7; ++h)
    for (int w = 0; w < 7; ++w)
      s += (float)in[(((long)n * 7 + h) * 7 + w) * 2048 + c];
  out[idx] = s * (1.0f / 49.0f);
}

// ---------------------------------------------------------------------------
// FC: [8,2048] @ [2048,1000] + bias -> [8,1000] fp32
// ---------------------------------------------------------------------------
__global__ void fc_kernel(const float* __restrict__ h, const float* __restrict__ w,
                          const float* __restrict__ b, float* __restrict__ out)
{
  int idx = blockIdx.x * blockDim.x + threadIdx.x;
  if (idx >= 8 * 1000) return;
  int j = idx % 1000;
  int n = idx / 1000;
  float acc = b[j];
  for (int c = 0; c < 2048; ++c)
    acc += h[n * 2048 + c] * w[(long)c * 1000 + j];
  out[idx] = acc;
}

// ---------------------------------------------------------------------------
// WMMA implicit-GEMM convolution, NHWC fp16, fp32 accumulate.
//   A: activations [N,Hi,Wi,Cin] fp16 (direct im2col gather, L2-resident)
//   Wt: repacked weights [Cout][KH*KW*Cin] fp16 -> double-buffered through LDS
//       with GLOBAL_LOAD_ASYNC_TO_LDS_B128 (ASYNCcnt pipeline).
//   Epilogue: *scale[c] + bias[c] (+ residual) (+ ReLU) -> NHWC fp16.
// TAPS = KH*KW (1 or 9). K = TAPS*Cin is always a multiple of 32.
// Wave tile = 32x32 (2x2 of 16x16), block = 8 waves -> 64(M) x 128(N).
// ---------------------------------------------------------------------------
template<int TAPS>
__global__ __launch_bounds__(256)
void conv_wmma_kernel(const _Float16* __restrict__ A,
                      const _Float16* __restrict__ Wt,
                      const float* __restrict__ scale,
                      const float* __restrict__ bias,
                      const _Float16* __restrict__ resid,
                      _Float16* __restrict__ O,
                      int NB, int Hi, int Wi, int Cin, int cinShift,
                      int Ho, int Wo, int Cout, int stride, int pad, int relu)
{
  // Double-buffered weight tile: 128 cout x 32 k fp16 = 8 KB per buffer.
  __shared__ _Float16 smemW[2][128 * 32];

  const int K = TAPS * Cin;
  const int M = NB * Ho * Wo;
  const int lane = threadIdx.x & 31;
  const int wave = threadIdx.x >> 5;
  const int wm = wave & 1;     // 2 M sub-tiles of 32
  const int wn = wave >> 1;    // 4 N sub-tiles of 32
  const int lr = lane & 15;    // row / col within 16
  const int lh = lane >> 4;    // lane-half selects K group
  const int mBase = blockIdx.x * 64  + wm * 32;
  const int nBase = blockIdx.y * 128 + wn * 32;
  const int cBase = blockIdx.y * 128;

  // Decompose the two output rows this lane sources for the A fragments.
  int an[2], aho[2], awo[2];
  const _Float16* abase[2];
  #pragma unroll
  for (int t = 0; t < 2; ++t) {
    int r  = mBase + t * 16 + lr;
    int rc = r < M ? r : (M - 1);
    awo[t] = rc % Wo; int q = rc / Wo;
    aho[t] = q % Ho;  an[t] = q / Ho;
    abase[t] = A + (((size_t)an[t] * Hi + (size_t)aho[t] * stride) * Wi
                    + (size_t)awo[t] * stride) * Cin;
  }

  // Asynchronously copy the block's 128x32 weight tile for K-step k0 into
  // smemW[buf].  512 16-byte chunks, 2 per thread.  ASYNCcnt-tracked, no
  // VGPR staging.
  auto stageW = [&](int buf, int k0) {
    #pragma unroll
    for (int it = 0; it < 2; ++it) {
      int q  = (int)threadIdx.x + it * 256;      // chunk id 0..511
      int c  = q >> 2;                           // cout row 0..127
      int ko = (q & 3) * 8;                      // k offset 0,8,16,24
      int cc = cBase + c; cc = cc < Cout ? cc : Cout - 1;
      const _Float16* g = Wt + (size_t)cc * K + k0 + ko;
      unsigned ldsOff = (unsigned)(uintptr_t)(&smemW[buf][c * 32 + ko]);
      asm volatile("global_load_async_to_lds_b128 %0, %1, off"
                   :: "v"(ldsOff), "v"((unsigned long long)(uintptr_t)g)
                   : "memory");
    }
  };

  v8f acc[2][2] = {};
  const int kgrpA = lh << 3;   // A lane-half: k offsets {0..7,16..23} / {8..15,24..31}
  const int kgrpB = lh << 3;   // B lane-half: k offsets {0..7,16..23} / {8..15,24..31}

  const int nStages = K >> 5;
  stageW(0, 0);

  for (int ks = 0; ks < nStages; ++ks) {
    const int k0  = ks << 5;
    const int cur = ks & 1;

    // Own async copies into smemW[cur] complete; barrier => everyone's are,
    // and all waves have finished reading the buffer we are about to refill.
    asm volatile("s_wait_asynccnt 0x0" ::: "memory");
    __syncthreads();
    if (ks + 1 < nStages) stageW(cur ^ 1, k0 + 32);

    v16h afrag[2], bfrag[2];
    // ---- A fragments: direct global im2col gather ----
    #pragma unroll
    for (int t = 0; t < 2; ++t) {
      #pragma unroll
      for (int h2 = 0; h2 < 2; ++h2) {
        const int kc = k0 + kgrpA + h2 * 16;
        v8h chunk;
        #pragma unroll
        for (int j = 0; j < 8; ++j) chunk[j] = (_Float16)0.f;
        if (TAPS == 1) {
          chunk = *(const v8h*)(abase[t] + kc);
        } else {
          // K order is (kh,kw,cin); Cin is a power of two, each aligned
          // 8-chunk stays inside one (kh,kw) tap.
          const int tap = kc >> cinShift;
          const int cin = kc & (Cin - 1);
          const int kh = tap / 3, kw = tap - kh * 3;
          const int hi = aho[t] * stride - pad + kh;
          const int wi = awo[t] * stride - pad + kw;
          if (hi >= 0 && hi < Hi && wi >= 0 && wi < Wi)
            chunk = *(const v8h*)(A + (((size_t)an[t] * Hi + hi) * Wi + wi) * Cin + cin);
        }
        #pragma unroll
        for (int j = 0; j < 8; ++j) afrag[t][h2 * 8 + j] = chunk[j];
      }
    }
    // ---- B fragments from LDS (ds_load_b128 x2 per tile) ----
    #pragma unroll
    for (int t = 0; t < 2; ++t) {
      const _Float16* bp = &smemW[cur][(wn * 32 + t * 16 + lr) * 32 + kgrpB];
      v8h b0 = *(const v8h*)bp;
      v8h b1 = *(const v8h*)(bp + 16);
      #pragma unroll
      for (int j = 0; j < 8; ++j) { bfrag[t][j] = b0[j]; bfrag[t][8 + j] = b1[j]; }
    }
    // ---- 2x2 matrix-core tiles ----
    #pragma unroll
    for (int mt = 0; mt < 2; ++mt)
      #pragma unroll
      for (int nt = 0; nt < 2; ++nt)
        acc[mt][nt] = __builtin_amdgcn_wmma_f32_16x16x32_f16(
            false, afrag[mt], false, bfrag[nt], (short)0, acc[mt][nt], false, false);
  }

  // Epilogue: D layout = lane {n=lr, m=lh*8+r} per VGPR r.
  #pragma unroll
  for (int nt = 0; nt < 2; ++nt) {
    const int c = nBase + nt * 16 + lr;
    if (c >= Cout) continue;
    const float sc = scale[c];
    const float bi = bias[c];
    #pragma unroll
    for (int mt = 0; mt < 2; ++mt) {
      #pragma unroll
      for (int r = 0; r < 8; ++r) {
        const int mrow = mBase + mt * 16 + lh * 8 + r;
        if (mrow < M) {
          const size_t oidx = (size_t)mrow * Cout + c;
          float vv = acc[mt][nt][r] * sc + bi;
          if (resid) vv += (float)resid[oidx];
          if (relu)  vv = vv > 0.f ? vv : 0.f;
          O[oidx] = (_Float16)vv;
        }
      }
    }
  }
}

// ===========================================================================
// Host orchestration
// ===========================================================================
extern "C" void kernel_launch(void* const* d_in, const int* in_sizes, int n_in,
                              void* d_out, int out_size, void* d_ws, size_t ws_size,
                              hipStream_t stream)
{
  (void)in_sizes; (void)n_in; (void)out_size; (void)ws_size;
  const int LAYERSc[4] = {3, 4, 6, 3};
  const int PLANESc[4] = {64, 128, 256, 512};
  const int STRIDEc[4] = {1, 2, 2, 2};

  // ---- walk flattened inputs in dict-insertion order ----
  int ix = 0;
  struct BNp { const float *g, *b, *m, *v; };
  const float* x       = (const float*)d_in[ix++];
  const float* stem_w  = (const float*)d_in[ix++];
  BNp stem_bn; stem_bn.g = (const float*)d_in[ix++]; stem_bn.b = (const float*)d_in[ix++];
  stem_bn.m = (const float*)d_in[ix++]; stem_bn.v = (const float*)d_in[ix++];

  struct SCp { const float* w; BNp bn; } sc[4];
  for (int li = 0; li < 4; ++li) {
    sc[li].w = (const float*)d_in[ix++];
    sc[li].bn.g = (const float*)d_in[ix++]; sc[li].bn.b = (const float*)d_in[ix++];
    sc[li].bn.m = (const float*)d_in[ix++]; sc[li].bn.v = (const float*)d_in[ix++];
  }
  struct Blk { const float* w1; BNp bn1; const float* w2; const float* b2; BNp bn2;
               const float* w3; BNp bn3; bool hasD; const float* dw; BNp dbn; };
  Blk blocks[16]; int nb = 0;
  for (int li = 0; li < 4; ++li)
    for (int b = 0; b < LAYERSc[li]; ++b) {
      Blk k;
      k.w1 = (const float*)d_in[ix++];
      k.bn1.g=(const float*)d_in[ix++]; k.bn1.b=(const float*)d_in[ix++];
      k.bn1.m=(const float*)d_in[ix++]; k.bn1.v=(const float*)d_in[ix++];
      k.w2 = (const float*)d_in[ix++];
      k.b2 = (const float*)d_in[ix++];
      k.bn2.g=(const float*)d_in[ix++]; k.bn2.b=(const float*)d_in[ix++];
      k.bn2.m=(const float*)d_in[ix++]; k.bn2.v=(const float*)d_in[ix++];
      k.w3 = (const float*)d_in[ix++];
      k.bn3.g=(const float*)d_in[ix++]; k.bn3.b=(const float*)d_in[ix++];
      k.bn3.m=(const float*)d_in[ix++]; k.bn3.v=(const float*)d_in[ix++];
      k.hasD = (b == 0);
      if (k.hasD) {
        k.dw = (const float*)d_in[ix++];
        k.dbn.g=(const float*)d_in[ix++]; k.dbn.b=(const float*)d_in[ix++];
        k.dbn.m=(const float*)d_in[ix++]; k.dbn.v=(const float*)d_in[ix++];
      } else { k.dw = nullptr; k.dbn = BNp{nullptr,nullptr,nullptr,nullptr}; }
      blocks[nb++] = k;
    }
  const float* fc_w = (const float*)d_in[ix++];
  const float* fc_b = (const float*)d_in[ix++];

  // ---- deterministic bump allocator over d_ws ----
  char* ws = (char*)d_ws;
  size_t off = 0;
  auto alloc = [&](size_t bytes) -> char* {
    char* p = ws + off;
    off += (bytes + 255) & ~(size_t)255;
    return p;
  };
  const size_t ACT = (size_t)8 * 112 * 112 * 64 * sizeof(_Float16);  // max activation (12.8 MB)
  _Float16* B0 = (_Float16*)alloc(ACT);
  _Float16* B1 = (_Float16*)alloc(ACT);
  _Float16* B2 = (_Float16*)alloc(ACT);
  _Float16* B3 = (_Float16*)alloc(ACT);
  _Float16* B4 = (_Float16*)alloc(ACT);
  _Float16* B5 = (_Float16*)alloc(ACT);
  float* pool = (float*)alloc((size_t)8 * 2048 * sizeof(float));

  // ---- conv runner: repack + fold + WMMA GEMM ----
  auto runConv = [&](const _Float16* in, _Float16* outp,
                     const float* wOIHW, const BNp& bn, const float* convBias,
                     int NB, int Hi, int Wi, int Cin, int KH,
                     int Ho, int Wo, int Cout, int stride, int pad,
                     const _Float16* resid, int relu) {
    const int K = KH * KH * Cin;
    _Float16* wr  = (_Float16*)alloc((size_t)Cout * K * sizeof(_Float16));
    float*    scl = (float*)alloc((size_t)Cout * sizeof(float));
    float*    bia = (float*)alloc((size_t)Cout * sizeof(float));
    {
      long total = (long)Cout * K;
      repack_w_kernel<<<(unsigned)((total + 255) / 256), 256, 0, stream>>>(wOIHW, wr, Cout, Cin, KH, KH);
    }
    fold_bn_kernel<<<(Cout + 255) / 256, 256, 0, stream>>>(bn.g, bn.b, bn.m, bn.v, convBias, scl, bia, Cout);
    const int M = NB * Ho * Wo;
    dim3 grid((M + 63) / 64, (Cout + 127) / 128);
    int cinShift = 31 - __builtin_clz((unsigned)Cin);
    if (KH == 1)
      conv_wmma_kernel<1><<<grid, 256, 0, stream>>>(in, wr, scl, bia, resid, outp,
          NB, Hi, Wi, Cin, cinShift, Ho, Wo, Cout, stride, pad, relu);
    else
      conv_wmma_kernel<9><<<grid, 256, 0, stream>>>(in, wr, scl, bia, resid, outp,
          NB, Hi, Wi, Cin, cinShift, Ho, Wo, Cout, stride, pad, relu);
  };

  // ---- stem + maxpool ----
  {
    float* sscl = (float*)alloc(64 * sizeof(float));
    float* sbia = (float*)alloc(64 * sizeof(float));
    fold_bn_kernel<<<1, 64, 0, stream>>>(stem_bn.g, stem_bn.b, stem_bn.m, stem_bn.v, nullptr, sscl, sbia, 64);
    long total = (long)8 * 112 * 112 * 64;
    stem_kernel<<<(unsigned)((total + 255) / 256), 256, 0, stream>>>(x, stem_w, sscl, sbia, B0);
  }
  {
    long total = (long)8 * 56 * 56 * 64;
    maxpool_kernel<<<(unsigned)((total + 255) / 256), 256, 0, stream>>>(B0, B1);
  }

  // ---- stages ----
  _Float16* cur = B1;
  _Float16* other = B0;
  int H = 56, Cin = 64, bidx = 0;
  for (int li = 0; li < 4; ++li) {
    const int P = PLANESc[li], S = STRIDEc[li], C4 = P * 4;
    const int Hout = H / S;
    // stage-level shortcut from stage input
    runConv(cur, B5, sc[li].w, sc[li].bn, nullptr, 8, H, H, Cin, 1, Hout, Hout, C4, S, 0, nullptr, 0);
    for (int b = 0; b < LAYERSc[li]; ++b) {
      Blk& k = blocks[bidx++];
      const int s    = (b == 0) ? S : 1;
      const int hin  = (b == 0) ? H : Hout;
      const int cinB = (b == 0) ? Cin : C4;
      const int hout = hin / s;
      // conv1 1x1 + BN + ReLU
      runConv(cur, B2, k.w1, k.bn1, nullptr, 8, hin, hin, cinB, 1, hin, hin, P, 1, 0, nullptr, 1);
      // conv2 3x3 (bias folded) + BN + ReLU, stride s
      runConv(B2, B3, k.w2, k.bn2, k.b2, 8, hin, hin, P, 3, hout, hout, P, s, 1, nullptr, 1);
      // identity / downsample
      const _Float16* resid;
      if (k.hasD) {
        runConv(cur, B4, k.dw, k.dbn, nullptr, 8, hin, hin, cinB, 1, hout, hout, C4, s, 0, nullptr, 0);
        resid = B4;
      } else {
        resid = cur;
      }
      // conv3 1x1 + BN + residual + ReLU
      runConv(B3, other, k.w3, k.bn3, nullptr, 8, hout, hout, P, 1, hout, hout, C4, 1, 0, resid, 1);
      _Float16* tmp = cur; cur = other; other = tmp;
    }
    // stage output + stage-level shortcut (no relu)
    {
      long total = (long)8 * Hout * Hout * C4;
      add_kernel<<<(unsigned)((total + 255) / 256), 256, 0, stream>>>(cur, B5, other, total);
      _Float16* tmp = cur; cur = other; other = tmp;
    }
    H = Hout; Cin = C4;
  }

  // ---- avgpool + FC ----
  avgpool_kernel<<<(8 * 2048 + 255) / 256, 256, 0, stream>>>(cur, pool);
  fc_kernel<<<(8 * 1000 + 255) / 256, 256, 0, stream>>>(pool, fc_w, fc_b, (float*)d_out);
}
